// multi_task_baseline_70351564308757
// MI455X (gfx1250) — matrix operand order repaired and checked by reading, not compiled, not run on hardware
//
#include <hip/hip_runtime.h>

typedef __attribute__((ext_vector_type(16))) _Float16 v16h;
typedef __attribute__((ext_vector_type(8)))  float    v8f;

#define B_DIM 256
#define C_DIM 128
#define H_DIM 1024
#define I_DIM 512
#define O_DIM 128

// Build an A-matrix f16 fragment (16x32 layout) from two runs of 8 consecutive f32.
__device__ __forceinline__ v16h cvtA_f32(const float* __restrict__ p0,
                                         const float* __restrict__ p1) {
  const float4* a = (const float4*)p0;
  const float4* b = (const float4*)p1;
  float4 x0 = a[0], x1 = a[1], y0 = b[0], y1 = b[1];
  v16h f;
  f[0]  = (_Float16)x0.x; f[1]  = (_Float16)x0.y; f[2]  = (_Float16)x0.z; f[3]  = (_Float16)x0.w;
  f[4]  = (_Float16)x1.x; f[5]  = (_Float16)x1.y; f[6]  = (_Float16)x1.z; f[7]  = (_Float16)x1.w;
  f[8]  = (_Float16)y0.x; f[9]  = (_Float16)y0.y; f[10] = (_Float16)y0.z; f[11] = (_Float16)y0.w;
  f[12] = (_Float16)y1.x; f[13] = (_Float16)y1.y; f[14] = (_Float16)y1.z; f[15] = (_Float16)y1.w;
  return f;
}

// Build a B-matrix f16 fragment (32x16 layout) from 16 consecutive f32.
__device__ __forceinline__ v16h cvtB_f32(const float* __restrict__ p) {
  const float4* a = (const float4*)p;
  float4 x0 = a[0], x1 = a[1], x2 = a[2], x3 = a[3];
  v16h f;
  f[0]  = (_Float16)x0.x; f[1]  = (_Float16)x0.y; f[2]  = (_Float16)x0.z; f[3]  = (_Float16)x0.w;
  f[4]  = (_Float16)x1.x; f[5]  = (_Float16)x1.y; f[6]  = (_Float16)x1.z; f[7]  = (_Float16)x1.w;
  f[8]  = (_Float16)x2.x; f[9]  = (_Float16)x2.y; f[10] = (_Float16)x2.z; f[11] = (_Float16)x2.w;
  f[12] = (_Float16)x3.x; f[13] = (_Float16)x3.y; f[14] = (_Float16)x3.z; f[15] = (_Float16)x3.w;
  return f;
}

// A-matrix fragment directly from f16 memory (two 16-byte runs).
__device__ __forceinline__ v16h loadA_f16(const _Float16* __restrict__ p0,
                                          const _Float16* __restrict__ p1) {
  union { v16h h; uint4 u[2]; } u;
  u.u[0] = *(const uint4*)p0;
  u.u[1] = *(const uint4*)p1;
  return u.h;
}

// Layer 1: h[c,b,:] = relu(x[b,c,:] @ w1[c]^T + b1[c])   (f16 workspace, layout (C,B,H))
__global__ __launch_bounds__(256) void mlp_layer1(const float* __restrict__ x,
                                                  const float* __restrict__ w1,
                                                  const float* __restrict__ b1,
                                                  _Float16* __restrict__ hws) {
  const int c    = blockIdx.z;
  const int n0   = blockIdx.x * 128;       // H tile base
  const int m0   = blockIdx.y * 128;       // B tile base
  const int lane = threadIdx.x & 31;
  const int wid  = threadIdx.x >> 5;
  const int wn   = wid & 1;                // 2 waves across N (64 each)
  const int wm   = wid >> 1;               // 4 waves across M (32 each)
  const int lr   = lane & 15;
  const int sel  = lane >> 4;

  v8f acc[2][4] = {};

  const float* xrow[2];
#pragma unroll
  for (int mi = 0; mi < 2; ++mi) {
    const int b = m0 + wm * 32 + mi * 16 + lr;
    xrow[mi] = x + ((size_t)b * C_DIM + c) * I_DIM;
  }
  const float* wrow[4];
#pragma unroll
  for (int ni = 0; ni < 4; ++ni) {
    const int hh = n0 + wn * 64 + ni * 16 + lr;
    wrow[ni] = w1 + ((size_t)c * H_DIM + hh) * I_DIM;
  }

  for (int kk = 0; kk < I_DIM; kk += 32) {
    v16h af[2], bf[4];
    const int ka = kk + sel * 16;
    const int kb = kk + sel * 16;
#pragma unroll
    for (int mi = 0; mi < 2; ++mi) af[mi] = cvtA_f32(xrow[mi] + ka, xrow[mi] + ka + 8);
#pragma unroll
    for (int ni = 0; ni < 4; ++ni) bf[ni] = cvtB_f32(wrow[ni] + kb);
#pragma unroll
    for (int mi = 0; mi < 2; ++mi)
#pragma unroll
      for (int ni = 0; ni < 4; ++ni)
        acc[mi][ni] = __builtin_amdgcn_wmma_f32_16x16x32_f16(
            false, af[mi], false, bf[ni], (short)0, acc[mi][ni], false, false);
  }

#pragma unroll
  for (int ni = 0; ni < 4; ++ni) {
    const int hh = n0 + wn * 64 + ni * 16 + lr;
    const float bias = b1[(size_t)c * H_DIM + hh];
#pragma unroll
    for (int mi = 0; mi < 2; ++mi) {
#pragma unroll
      for (int r = 0; r < 8; ++r) {
        const int b = m0 + wm * 32 + mi * 16 + r + sel * 8;
        float v = acc[mi][ni][r] + bias;
        v = v > 0.0f ? v : 0.0f;
        hws[((size_t)c * B_DIM + b) * H_DIM + hh] = (_Float16)v;
      }
    }
  }
}

// Layer 2: lg[c,b,o] = h[c,b,:] @ w2[c]^T + b2[c]   (coalesced (C,B,O) staging layout)
__global__ __launch_bounds__(256) void mlp_layer2(const _Float16* __restrict__ hws,
                                                  const float* __restrict__ w2,
                                                  const float* __restrict__ b2,
                                                  float* __restrict__ lg) {
  const int c    = blockIdx.z;
  const int n0   = blockIdx.x * 128;       // O tile base (single tile)
  const int m0   = blockIdx.y * 128;       // B tile base
  const int lane = threadIdx.x & 31;
  const int wid  = threadIdx.x >> 5;
  const int wn   = wid & 1;
  const int wm   = wid >> 1;
  const int lr   = lane & 15;
  const int sel  = lane >> 4;

  v8f acc[2][4] = {};

  const _Float16* arow[2];
#pragma unroll
  for (int mi = 0; mi < 2; ++mi) {
    const int b = m0 + wm * 32 + mi * 16 + lr;
    arow[mi] = hws + ((size_t)c * B_DIM + b) * H_DIM;
  }
  const float* wrow[4];
#pragma unroll
  for (int ni = 0; ni < 4; ++ni) {
    const int o = n0 + wn * 64 + ni * 16 + lr;
    wrow[ni] = w2 + ((size_t)c * O_DIM + o) * H_DIM;
  }

  for (int kk = 0; kk < H_DIM; kk += 32) {
    v16h af[2], bf[4];
    const int ka = kk + sel * 16;
    const int kb = kk + sel * 16;
#pragma unroll
    for (int mi = 0; mi < 2; ++mi) af[mi] = loadA_f16(arow[mi] + ka, arow[mi] + ka + 8);
#pragma unroll
    for (int ni = 0; ni < 4; ++ni) bf[ni] = cvtB_f32(wrow[ni] + kb);
#pragma unroll
    for (int mi = 0; mi < 2; ++mi)
#pragma unroll
      for (int ni = 0; ni < 4; ++ni)
        acc[mi][ni] = __builtin_amdgcn_wmma_f32_16x16x32_f16(
            false, af[mi], false, bf[ni], (short)0, acc[mi][ni], false, false);
  }

  // Coalesced store: lanes 0-15 cover 16 consecutive o (64B run), lanes 16-31 likewise at b+8.
#pragma unroll
  for (int ni = 0; ni < 4; ++ni) {
    const int o = n0 + wn * 64 + ni * 16 + lr;
    const float bias = b2[(size_t)c * O_DIM + o];
#pragma unroll
    for (int mi = 0; mi < 2; ++mi) {
#pragma unroll
      for (int r = 0; r < 8; ++r) {
        const int b = m0 + wm * 32 + mi * 16 + r + sel * 8;
        lg[((size_t)c * B_DIM + b) * O_DIM + o] = acc[mi][ni][r] + bias;
      }
    }
  }
}

// Transpose (C,B,O) -> (B,O,C): for each b, a 128x128 (c,o)->(o,c) tile transpose via LDS.
__global__ __launch_bounds__(256) void transpose_oc(const float* __restrict__ lg,
                                                    float* __restrict__ out) {
  __shared__ float tile[32][33];          // +1 pad: conflict-free transposed reads
  const int b  = blockIdx.z;
  const int c0 = blockIdx.x * 32;
  const int o0 = blockIdx.y * 32;
  const int tx = threadIdx.x;             // 0..31
  const int ty = threadIdx.y;             // 0..7
#pragma unroll
  for (int i = 0; i < 4; ++i) {
    const int cc = ty + i * 8;
    tile[cc][tx] = lg[((size_t)(c0 + cc) * B_DIM + b) * O_DIM + (o0 + tx)];
  }
  __syncthreads();
#pragma unroll
  for (int i = 0; i < 4; ++i) {
    const int oo = ty + i * 8;
    out[((size_t)b * O_DIM + (o0 + oo)) * C_DIM + (c0 + tx)] = tile[tx][oo];
  }
}

extern "C" void kernel_launch(void* const* d_in, const int* in_sizes, int n_in,
                              void* d_out, int out_size, void* d_ws, size_t ws_size,
                              hipStream_t stream) {
  (void)in_sizes; (void)n_in; (void)out_size; (void)ws_size;
  const float* x  = (const float*)d_in[0];
  const float* w1 = (const float*)d_in[1];
  const float* b1 = (const float*)d_in[2];
  const float* w2 = (const float*)d_in[3];
  const float* b2 = (const float*)d_in[4];
  float* out = (float*)d_out;

  // Workspace: [0, 64MiB) h as f16 (C,B,H); [64MiB, 80MiB) logits f32 (C,B,O)
  _Float16* hws = (_Float16*)d_ws;
  float*    lg  = (float*)((char*)d_ws + (size_t)C_DIM * B_DIM * H_DIM * sizeof(_Float16));

  dim3 blk(256, 1, 1);
  dim3 g1(H_DIM / 128, B_DIM / 128, C_DIM);   // (8, 2, 128)
  dim3 g2(O_DIM / 128, B_DIM / 128, C_DIM);   // (1, 2, 128)
  mlp_layer1<<<g1, blk, 0, stream>>>(x, w1, b1, hws);
  mlp_layer2<<<g2, blk, 0, stream>>>(hws, w2, b2, lg);

  dim3 tblk(32, 8, 1);
  dim3 g3(C_DIM / 32, O_DIM / 32, B_DIM);     // (4, 4, 256)
  transpose_oc<<<g3, tblk, 0, stream>>>(lg, out);
}
